// GeneratorModel_61813169324133
// MI455X (gfx1250) — compile-verified, matrix operand order checked by reading, not run on hardware
//
#include <hip/hip_runtime.h>
#include <hip/hip_bf16.h>

typedef __attribute__((ext_vector_type(16))) _Float16 v16h;
typedef __attribute__((ext_vector_type(8)))  float    v8f;

#define NSAMP   500000
#define MAXLEN  20
#define HD      8      // hidden
#define VD      28     // vocab
#define GD      32     // 4*H gates
#define WPB     8      // waves per block

// ---------- small helpers ----------
__device__ __forceinline__ unsigned mix3(unsigned a, unsigned b, unsigned c) {
  unsigned h = a * 0x9E3779B1u + b * 0x85EBCA77u + c * 0xC2B2AE3Du + 0x27220A95u;
  h ^= h >> 16; h *= 0x7FEB352Du;
  h ^= h >> 15; h *= 0x846CA68Bu;
  h ^= h >> 16;
  return h;
}
__device__ __forceinline__ float gumbel(unsigned s, unsigned k, unsigned v) {
  float u = ((float)mix3(s, k, v) + 0.5f) * (1.0f / 4294967296.0f); // (0,1)
  return -__logf(-__logf(u));
}
__device__ __forceinline__ float sigm(float x) { return 1.0f / (1.0f + __expf(-x)); }
__device__ __forceinline__ float tanhf_fast(float x) {
  float e = __expf(2.0f * x);
  return (e - 1.0f) / (e + 1.0f);
}
__device__ __forceinline__ v8f wmma16(v16h a, v16h b, v8f c) {
  // D = A(16x32 f16) * B(32x16 f16) + C(16x16 f32)
  return __builtin_amdgcn_wmma_f32_16x16x32_f16(false, a, false, b, (short)0, c, false, false);
}

// One LSTM cell update. A = h in WMMA-A layout (f16, K=0..7 valid).
// cat[r] = input token for sample (8*grp + r) of this wave's 16-sample tile.
// State hs/cs live on lanes with (lane&15) < 8: lane j, reg r -> h[8*grp+r][j].
__device__ __forceinline__ void cell_update(
    v16h A, const int* cat, float* hs, float* cs,
    v16h Bhh0, v16h Bhh1, float bias0, float bias1,
    const float* s_Wih, int col)
{
  v8f C0, C1;
#pragma unroll
  for (int r = 0; r < HD; ++r) {
    // gates = W_ih[:,cat] + bias  (one-hot input term folded into C operand)
    C0[r] = bias0 + s_Wih[col * VD + cat[r]];
    C1[r] = bias1 + s_Wih[(16 + col) * VD + cat[r]];
  }
  v8f D0 = wmma16(A, Bhh0, C0);   // gates 0..15  (i | f)
  v8f D1 = wmma16(A, Bhh1, C1);   // gates 16..31 (g | o)
#pragma unroll
  for (int r = 0; r < HD; ++r) {
    float iv = sigm(D0[r]);
    float fv = sigm(__shfl_xor(D0[r], 8, 32));   // pair lane j with j+8 (f gate)
    float gv = tanhf_fast(D1[r]);
    float ov = sigm(__shfl_xor(D1[r], 8, 32));   // o gate
    float cn = fv * cs[r] + iv * gv;
    cs[r] = cn;
    hs[r] = ov * tanhf_fast(cn);
  }
}

extern "C" __global__ void __launch_bounds__(256)
lstm_generate(const float* __restrict__ W_ih, const float* __restrict__ W_hh,
              const float* __restrict__ b_ih, const float* __restrict__ b_hh,
              const float* __restrict__ W_fc, const float* __restrict__ b_fc,
              float* __restrict__ out)
{
  __shared__ float s_Wih[GD * VD];          // 3.5 KB: input weight for per-token gather
  __shared__ float s_hT[WPB][16 * HD];      // per-wave h transpose staging
  __shared__ int   s_cat[WPB][16];          // per-wave sampled tokens

  __builtin_prefetch(W_ih, 0, 0);
  for (int i = threadIdx.x; i < GD * VD; i += blockDim.x) s_Wih[i] = W_ih[i];
  __syncthreads();

  const int wave = threadIdx.x >> 5;
  const int lane = threadIdx.x & 31;
  const unsigned gwave = blockIdx.x * WPB + wave;
  if (gwave >= (NSAMP / 16)) return;        // whole-wave exit: EXEC stays all-ones
  const unsigned tile_base = gwave * 16;    // 16 samples per wave

  const int grp = lane >> 4;                // sample half (rows 0-7 vs 8-15)
  const int col = lane & 15;                // gate/logit column

  // ---- constant B tiles (f16), K=0..7 valid, lanes 16-31 hold K=16..31 == 0
  v16h Bhh0 = {}; v16h Bhh1 = {}; v16h Bfc0 = {}; v16h Bfc1 = {};
  if (grp == 0) {
#pragma unroll
    for (int kk = 0; kk < HD; ++kk) {
      Bhh0[kk] = (_Float16)W_hh[col * HD + kk];
      Bhh1[kk] = (_Float16)W_hh[(16 + col) * HD + kk];
      Bfc0[kk] = (_Float16)W_fc[col * HD + kk];
      Bfc1[kk] = (16 + col < VD) ? (_Float16)W_fc[(16 + col) * HD + kk]
                                 : (_Float16)0.0f;
    }
  }
  const float bias0 = b_ih[col]      + b_hh[col];
  const float bias1 = b_ih[16 + col] + b_hh[16 + col];
  const float bf0 = b_fc[col];
  const float bf1 = (16 + col < VD) ? b_fc[16 + col] : 0.0f;

  float hs[HD], cs[HD];
  int   cat[HD];
#pragma unroll
  for (int r = 0; r < HD; ++r) { hs[r] = 0.0f; cs[r] = 0.0f; cat[r] = 0; }

  // ---- initial cell: h = 0, x = one-hot(START_IDX=0)
  {
    v16h A = {};
    cell_update(A, cat, hs, cs, Bhh0, Bhh1, bias0, bias1, s_Wih, col);
  }

  for (int step = 0; step < MAXLEN; ++step) {
    // ---- transpose h (gate layout) -> WMMA A layout via per-wave LDS
    if (col < HD) {
#pragma unroll
      for (int r = 0; r < HD; ++r)
        s_hT[wave][(grp * 8 + r) * HD + col] = hs[r];
    }
    asm volatile("s_wait_dscnt 0x0" ::: "memory");
    v16h A = {};
    if (grp == 0) {
#pragma unroll
      for (int kk = 0; kk < HD; ++kk)
        A[kk] = (_Float16)s_hT[wave][col * HD + kk];   // row = lane, K = kk
    }

    // ---- logits = h @ W_fc.T + b_fc  (two 16-col WMMA tiles)
    v8f Cf0, Cf1;
#pragma unroll
    for (int r = 0; r < HD; ++r) { Cf0[r] = bf0; Cf1[r] = bf1; }
    v8f L0 = wmma16(A, Bfc0, Cf0);
    v8f L1 = wmma16(A, Bfc1, Cf1);

    // ---- Gumbel-argmax categorical sampling, per sample (reg) across 16 lanes
#pragma unroll
    for (int r = 0; r < HD; ++r) {
      unsigned smp = tile_base + grp * 8 + r;
      float v0 = L0[r] + gumbel(smp, (unsigned)step, (unsigned)col);
      float v1 = (16 + col < VD)
                   ? (L1[r] + gumbel(smp, (unsigned)step, (unsigned)(16 + col)))
                   : -3.0e38f;
      float best; int bidx;
      if (v1 > v0) { best = v1; bidx = 16 + col; } else { best = v0; bidx = col; }
#pragma unroll
      for (int m = 1; m < 16; m <<= 1) {
        float ob = __shfl_xor(best, m, 32);
        int   oi = __shfl_xor(bidx, m, 32);
        if (ob > best || (ob == best && oi < bidx)) { best = ob; bidx = oi; }
      }
      cat[r] = bidx;   // uniform across the 16-lane group
    }

    // ---- publish tokens; write one-hot rows (7 aligned b128 stores / sample)
    if (col == 0) {
#pragma unroll
      for (int r = 0; r < HD; ++r) s_cat[wave][grp * 8 + r] = cat[r];
    }
    asm volatile("s_wait_dscnt 0x0" ::: "memory");
    if (grp == 0) {
      int cm = s_cat[wave][col];
      unsigned smp = tile_base + col;
      float4* dst = (float4*)(out + ((size_t)smp * MAXLEN + step) * VD);
#pragma unroll
      for (int q = 0; q < VD / 4; ++q) {
        float4 w;
        w.x = (4 * q + 0 == cm) ? 1.0f : 0.0f;
        w.y = (4 * q + 1 == cm) ? 1.0f : 0.0f;
        w.z = (4 * q + 2 == cm) ? 1.0f : 0.0f;
        w.w = (4 * q + 3 == cm) ? 1.0f : 0.0f;
        dst[q] = w;
      }
    }

    // ---- LSTM cell with sampled token
    cell_update(A, cat, hs, cs, Bhh0, Bhh1, bias0, bias1, s_Wih, col);
  }
}

extern "C" void kernel_launch(void* const* d_in, const int* in_sizes, int n_in,
                              void* d_out, int out_size, void* d_ws, size_t ws_size,
                              hipStream_t stream) {
  // setup_inputs order: samples, max_len, W_ih, W_hh, b_ih, b_hh, W_fc, b_fc
  const float* W_ih = (const float*)d_in[2];
  const float* W_hh = (const float*)d_in[3];
  const float* b_ih = (const float*)d_in[4];
  const float* b_hh = (const float*)d_in[5];
  const float* W_fc = (const float*)d_in[6];
  const float* b_fc = (const float*)d_in[7];
  float* out = (float*)d_out;

  const int waves  = NSAMP / 16;            // 31250 (exact)
  const int blocks = (waves + WPB - 1) / WPB; // 3907
  lstm_generate<<<blocks, 256, 0, stream>>>(W_ih, W_hh, b_ih, b_hh,
                                            W_fc, b_fc, out);
}
